// GlobalSelfAttentionHead_32959579030103
// MI455X (gfx1250) — matrix operand order repaired and checked by reading, not compiled, not run on hardware
//
#include <hip/hip_runtime.h>
#include <cstdint>

// ---------------------------------------------------------------------------
// Problem constants (reference: B=2, C=1024, CI=512, T=32, H=W=14, N=6272)
// ---------------------------------------------------------------------------
#define BSZ   2
#define CDIM  1024
#define CIDIM 512
#define TDIM  32
#define NDIM  6272   // 49 * 128

typedef __bf16 bf16_t;
typedef __bf16 v8bf  __attribute__((ext_vector_type(8)));
typedef __bf16 v16bf __attribute__((ext_vector_type(16)));
typedef float  v8f   __attribute__((ext_vector_type(8)));
typedef int    v4i   __attribute__((ext_vector_type(4)));

// CDNA5 async global->LDS path (ASYNCcnt), guarded so compile can't regress.
// Probe round 2 showed the builtin exists and takes v4i pointers
// (global-AS source, LDS-AS dest), matching llvm.amdgcn.global.load.async.*.
#if defined(__gfx1250__) && __has_builtin(__builtin_amdgcn_global_load_async_to_lds_b128)
#define HAVE_ASYNC_LDS 1
typedef __attribute__((address_space(1))) v4i* gv4i_p;
typedef __attribute__((address_space(3))) v4i* lv4i_p;
#if __has_builtin(__builtin_amdgcn_s_wait_asynccnt)
#define WAIT_ASYNC(n) __builtin_amdgcn_s_wait_asynccnt(n)
#else
#define WAIT_ASYNC(n) asm volatile("s_wait_asynccnt %0" ::"i"(n))
#endif
#else
#define HAVE_ASYNC_LDS 0
#define WAIT_ASYNC(n) ((void)0)
#endif

static __device__ inline v16bf make16(v8bf lo, v8bf hi) {
  v16bf r;
#pragma unroll
  for (int i = 0; i < 8; ++i) { r[i] = lo[i]; r[i + 8] = hi[i]; }
  return r;
}

// ---------------------------------------------------------------------------
// Generic WMMA bf16 GEMM:  Out(M x Nn) = A(M x K, rowmajor) * Bt(Nn x K)^T
// lda = ldb = K, ldOut = Nn.  Block tile 128x128, 128 threads = 4 waves
// (wave32), wave tile 64x64 = 4x4 v_wmma_f32_16x16x32_bf16 per K-step of 32.
// Double-buffered LDS staging via GLOBAL_LOAD_ASYNC_TO_LDS_B128 when present.
// EPI: 0 = bf16 out (+bias[col] if non-null)
//      1 = bf16 out + bias[row]
//      2 = f32 out, scaled
//      3 = f32 out + bias[row] + residual (same layout as Out)
// ---------------------------------------------------------------------------
template <int EPI>
__global__ void __launch_bounds__(128)
wmma_gemm(const bf16_t* __restrict__ A,  long long sA,
          const bf16_t* __restrict__ Bt, long long sB,
          void* __restrict__ Out,        long long sO,
          int M, int Nn, int K,
          const float* __restrict__ bias,
          const float* __restrict__ resid, long long sR,
          float scale)
{
  __shared__ bf16_t As[2][128][32];
  __shared__ bf16_t Bs[2][128][32];

  const int t    = threadIdx.x;            // 0..127
  const int lane = t & 31;
  const int wave = t >> 5;                 // 0..3
  const int wy   = wave >> 1;              // 0..1 : 64-row slab in M
  const int wx   = wave & 1;               // 0..1 : 64-col slab in N
  const int lrow = lane & 15;
  const bool hi  = lane >= 16;

  const int blockM = blockIdx.y * 128;
  const int blockN = blockIdx.x * 128;
  const int batch  = blockIdx.z;

  // staging: each thread owns one full 32-element (64B) row of each tile
  const bf16_t* gA = A  + (size_t)batch * sA + (size_t)(blockM + t) * K;
  const bf16_t* gB = Bt + (size_t)batch * sB + (size_t)(blockN + t) * K;

  v8f acc[4][4];
#pragma unroll
  for (int i = 0; i < 4; ++i)
#pragma unroll
    for (int j = 0; j < 4; ++j)
#pragma unroll
      for (int r = 0; r < 8; ++r) acc[i][j][r] = 0.0f;

  // ---- stage one 128x32 A tile + 128x32 B tile into LDS buffer `buf` ----
  auto issue_stage = [&](int k0, int buf) {
#if HAVE_ASYNC_LDS
#pragma unroll
    for (int c = 0; c < 4; ++c)
      __builtin_amdgcn_global_load_async_to_lds_b128(
          (gv4i_p)(gA + k0 + c * 8), (lv4i_p)&As[buf][t][c * 8], 0, 0);
#pragma unroll
    for (int c = 0; c < 4; ++c)
      __builtin_amdgcn_global_load_async_to_lds_b128(
          (gv4i_p)(gB + k0 + c * 8), (lv4i_p)&Bs[buf][t][c * 8], 0, 0);
#else
    __builtin_prefetch((const void*)(gA + k0 + 32), 0, 3);
    __builtin_prefetch((const void*)(gB + k0 + 32), 0, 3);
#pragma unroll
    for (int c = 0; c < 4; ++c) {
      v8bf a = *(const v8bf*)(gA + k0 + c * 8);
      *(v8bf*)&As[buf][t][c * 8] = a;
    }
#pragma unroll
    for (int c = 0; c < 4; ++c) {
      v8bf b = *(const v8bf*)(gB + k0 + c * 8);
      *(v8bf*)&Bs[buf][t][c * 8] = b;
    }
#endif
  };

  // ---- 16 WMMAs on buffer `buf` -----------------------------------------
  auto compute = [&](int buf) {
    // B fragments: B is (32 x 16) K-major, k order chosen here (matches A); lane n = lrow,
    // lanes 0-15 hold K=0..15, lanes 16-31 hold K=16..31 (contiguous in row)
    v16bf bfrag[4];
#pragma unroll
    for (int nt = 0; nt < 4; ++nt) {
      const int brow = wx * 64 + nt * 16 + lrow;
      const int bk   = hi ? 16 : 0;
      bfrag[nt] = make16(*(const v8bf*)&Bs[buf][brow][bk],
                         *(const v8bf*)&Bs[buf][brow][bk + 8]);
    }
    // A fragments: A is (16 x 32), k order chosen here (matches B); lane m = lrow,
    // lanes 0-15 hold K=0..15, lanes 16-31 hold K=16..31 (contiguous in row)
#pragma unroll
    for (int mt = 0; mt < 4; ++mt) {
      const int arow = wy * 64 + mt * 16 + lrow;
      const int ak   = hi ? 16 : 0;
      v16bf afrag = make16(*(const v8bf*)&As[buf][arow][ak],
                           *(const v8bf*)&As[buf][arow][ak + 8]);
#pragma unroll
      for (int nt = 0; nt < 4; ++nt) {
        acc[mt][nt] = __builtin_amdgcn_wmma_f32_16x16x32_bf16(
            false, afrag, false, bfrag[nt], (short)0, acc[mt][nt], false, false);
      }
    }
  };

  // ---- double-buffered main loop (last tile peeled: constant wait imms) --
  int buf = 0;
  issue_stage(0, 0);
  for (int k0 = 0; k0 < K - 32; k0 += 32) {
    issue_stage(k0 + 32, buf ^ 1);
    WAIT_ASYNC(8);          // oldest 8 async ops (= current buffer) complete
    __syncthreads();
    compute(buf);
    __syncthreads();
    buf ^= 1;
  }
  WAIT_ASYNC(0);
  __syncthreads();
  compute(buf);

  // Epilogue.  C/D layout: VGPR r -> M = r (+8 for lanes 16-31), N = lane&15.
  const int rowBase = blockM + wy * 64 + (hi ? 8 : 0);
  const int col0    = blockN + wx * 64 + lrow;
#pragma unroll
  for (int mt = 0; mt < 4; ++mt) {
#pragma unroll
    for (int nt = 0; nt < 4; ++nt) {
      const int col = col0 + nt * 16;
#pragma unroll
      for (int r = 0; r < 8; ++r) {
        const int row = rowBase + mt * 16 + r;
        float v = acc[mt][nt][r] * scale;
        const size_t oidx = (size_t)batch * sO + (size_t)row * Nn + col;
        if (EPI == 0) {
          if (bias) v += bias[col];
          ((bf16_t*)Out)[oidx] = (bf16_t)v;
        } else if (EPI == 1) {
          v += bias[row];
          ((bf16_t*)Out)[oidx] = (bf16_t)v;
        } else if (EPI == 2) {
          ((float*)Out)[oidx] = v;
        } else {
          v += bias[row] + resid[(size_t)batch * sR + (size_t)row * Nn + col];
          ((float*)Out)[oidx] = v;
        }
      }
    }
  }
}

// ---------------------------------------------------------------------------
// Prep kernels
// ---------------------------------------------------------------------------
__global__ void cvt_f32_bf16(const float* __restrict__ in, bf16_t* __restrict__ out,
                             long long n) {
  long long i = (long long)blockIdx.x * blockDim.x + threadIdx.x;
  if (i < n) out[i] = (bf16_t)in[i];
}

// Xt[b][n][c] = x[b][c][n]  (bf16), indexed by output for coalesced writes
__global__ void build_xt(const float* __restrict__ x, bf16_t* __restrict__ xt) {
  long long i = (long long)blockIdx.x * blockDim.x + threadIdx.x;
  const long long total = (long long)BSZ * NDIM * CDIM;
  if (i >= total) return;
  int c = (int)(i % CDIM);
  long long bn = i / CDIM;
  int n = (int)(bn % NDIM);
  int b = (int)(bn / NDIM);
  xt[i] = (bf16_t)x[((long long)b * CDIM + c) * NDIM + n];
}

// Row softmax over N=6272 fp32 logits -> bf16 probabilities.
// One 256-thread block per (batch,row); the fp32 row stays cache-resident
// (per-batch S = 157 MB, inside the 192 MB L2).
__global__ void __launch_bounds__(256)
softmax_rows(const float* __restrict__ S, bf16_t* __restrict__ P) {
  const float* s = S + (size_t)blockIdx.x * NDIM;
  bf16_t* p = P + (size_t)blockIdx.x * NDIM;
  __shared__ float red[256];
  const int t = threadIdx.x;

  float mx = -3.4e38f;
  for (int j = t; j < NDIM; j += 256) mx = fmaxf(mx, s[j]);
  red[t] = mx;
  __syncthreads();
  for (int o = 128; o > 0; o >>= 1) {
    if (t < o) red[t] = fmaxf(red[t], red[t + o]);
    __syncthreads();
  }
  mx = red[0];
  __syncthreads();

  float sum = 0.0f;
  for (int j = t; j < NDIM; j += 256) sum += __expf(s[j] - mx);
  red[t] = sum;
  __syncthreads();
  for (int o = 128; o > 0; o >>= 1) {
    if (t < o) red[t] += red[t + o];
    __syncthreads();
  }
  const float inv = 1.0f / red[0];
  for (int j = t; j < NDIM; j += 256) p[j] = (bf16_t)(__expf(s[j] - mx) * inv);
}

// ---------------------------------------------------------------------------
// Host launcher
// ---------------------------------------------------------------------------
extern "C" void kernel_launch(void* const* d_in, const int* in_sizes, int n_in,
                              void* d_out, int out_size, void* d_ws, size_t ws_size,
                              hipStream_t stream) {
  (void)in_sizes; (void)n_in; (void)out_size; (void)ws_size;

  const float* x  = (const float*)d_in[0];
  const float* Wk = (const float*)d_in[1];
  const float* bk = (const float*)d_in[2];
  const float* Wq = (const float*)d_in[3];
  const float* bq = (const float*)d_in[4];
  const float* Wv = (const float*)d_in[5];
  const float* bv = (const float*)d_in[6];
  const float* Wo = (const float*)d_in[7];
  const float* bo = (const float*)d_in[8];
  float* out = (float*)d_out;

  char* ws = (char*)d_ws;
  size_t off = 0;
  auto carve = [&](size_t bytes) -> char* {
    char* p = ws + off;
    off += (bytes + 255) & ~(size_t)255;
    return p;
  };

  bf16_t* xt  = (bf16_t*)carve((size_t)BSZ * NDIM * CDIM * 2);   // Xt  (B,N,C)
  bf16_t* wkb = (bf16_t*)carve((size_t)CIDIM * CDIM * 2);        // Wk  (CI,C)
  bf16_t* wqb = (bf16_t*)carve((size_t)CIDIM * CDIM * 2);
  bf16_t* wvb = (bf16_t*)carve((size_t)CIDIM * CDIM * 2);
  bf16_t* wob = (bf16_t*)carve((size_t)CDIM * CIDIM * 2);        // Wo  (C,CI)
  bf16_t* kt  = (bf16_t*)carve((size_t)BSZ * NDIM * CIDIM * 2);  // Kt  (B,N,CI)
  bf16_t* qt  = (bf16_t*)carve((size_t)BSZ * NDIM * CIDIM * 2);  // Qt  (B,N,CI)
  bf16_t* vn  = (bf16_t*)carve((size_t)BSZ * CIDIM * NDIM * 2);  // Vn  (B,CI,N)
  bf16_t* yt  = (bf16_t*)carve((size_t)BSZ * NDIM * CIDIM * 2);  // Yt  (B,N,CI)
  float*  sb  = (float*) carve((size_t)BSZ * NDIM * NDIM * 4);   // S   (B,N,N) f32
  bf16_t* pb  = (bf16_t*)carve((size_t)BSZ * NDIM * NDIM * 2);   // P   (B,N,N) bf16

  // --- prep: bf16 weights + transposed activations ---
  {
    long long nw = (long long)CIDIM * CDIM;
    int g = (int)((nw + 255) / 256);
    cvt_f32_bf16<<<g, 256, 0, stream>>>(Wk, wkb, nw);
    cvt_f32_bf16<<<g, 256, 0, stream>>>(Wq, wqb, nw);
    cvt_f32_bf16<<<g, 256, 0, stream>>>(Wv, wvb, nw);
    cvt_f32_bf16<<<g, 256, 0, stream>>>(Wo, wob, nw);
    long long nx = (long long)BSZ * NDIM * CDIM;
    build_xt<<<(int)((nx + 255) / 256), 256, 0, stream>>>(x, xt);
  }

  const long long sXT  = (long long)NDIM * CDIM;
  const long long sKT  = (long long)NDIM * CIDIM;
  const long long sVN  = (long long)CIDIM * NDIM;
  const long long sS   = (long long)NDIM * NDIM;
  const long long sOUT = (long long)CDIM * NDIM;
  dim3 blk(128);

  // Kt = Xt * Wk^T  (+bk along cols)        M=N, Nn=CI, K=C
  wmma_gemm<0><<<dim3(CIDIM / 128, NDIM / 128, BSZ), blk, 0, stream>>>(
      xt, sXT, wkb, 0, kt, sKT, NDIM, CIDIM, CDIM, bk, nullptr, 0, 1.0f);
  // Qt = Xt * Wq^T  (+bq along cols)
  wmma_gemm<0><<<dim3(CIDIM / 128, NDIM / 128, BSZ), blk, 0, stream>>>(
      xt, sXT, wqb, 0, qt, sKT, NDIM, CIDIM, CDIM, bq, nullptr, 0, 1.0f);
  // Vn = Wv * X     (+bv along rows)        M=CI, Nn=N, K=C  (Bt = Xt)
  wmma_gemm<1><<<dim3(NDIM / 128, CIDIM / 128, BSZ), blk, 0, stream>>>(
      wvb, 0, xt, sXT, vn, sVN, CIDIM, NDIM, CDIM, bv, nullptr, 0, 1.0f);
  // S = (Kt * Qt^T) / T   fp32             M=N, Nn=N, K=CI
  wmma_gemm<2><<<dim3(NDIM / 128, NDIM / 128, BSZ), blk, 0, stream>>>(
      kt, sKT, qt, sKT, sb, sS, NDIM, NDIM, CIDIM, nullptr, nullptr, 0,
      1.0f / (float)TDIM);
  // P = softmax_rows(S)  -> bf16
  softmax_rows<<<BSZ * NDIM, 256, 0, stream>>>(sb, pb);
  // Yt = P * Vn^T                           M=N, Nn=CI, K=N  (Bt = Vn)
  wmma_gemm<0><<<dim3(CIDIM / 128, NDIM / 128, BSZ), blk, 0, stream>>>(
      pb, sS, vn, sVN, yt, sKT, NDIM, CIDIM, NDIM, nullptr, nullptr, 0, 1.0f);
  // out = Wo * Y + bo + x  (fp32 to d_out)  M=C, Nn=N, K=CI  (Bt = Yt)
  wmma_gemm<3><<<dim3(NDIM / 128, CDIM / 128, BSZ), blk, 0, stream>>>(
      wob, 0, yt, sKT, out, sOUT, CDIM, NDIM, CIDIM, bo, x, sOUT, 1.0f);
}
